// fw_rnn_cell_45466523796025
// MI455X (gfx1250) — matrix operand order repaired and checked by reading, not compiled, not run on hardware
//
#include <hip/hip_runtime.h>
#include <hip/hip_bf16.h>

// Problem constants (match reference)
#define B_ 32
#define T_ 256
#define C_ 128
#define U_ 1024
#define E_LR_ 0.5f
#define LN_L_DECAY_ (-0.05129329f)   // ln(0.95)

typedef __bf16 bf16_t;
typedef __attribute__((ext_vector_type(4)))  bf16_t bf16x4;
typedef __attribute__((ext_vector_type(8)))  bf16_t bf16x8;
typedef __attribute__((ext_vector_type(16))) bf16_t bf16x16;
typedef __attribute__((ext_vector_type(8)))  float  floatx8;

typedef __attribute__((ext_vector_type(4))) unsigned int u32x4;
typedef __attribute__((ext_vector_type(4))) int          i32x4;
typedef __attribute__((ext_vector_type(8))) int          i32x8;

union FragU { bf16x8 h[2]; bf16x16 v; };

#if __has_builtin(__builtin_amdgcn_tensor_load_to_lds) && \
    __has_builtin(__builtin_amdgcn_s_wait_tensorcnt)
#define USE_TDM 1
#else
#define USE_TDM 0
#endif

// ---------------------------------------------------------------------------
// TDM: DMA a contiguous 32 KB region (16384 bf16 elements) from global to LDS.
// D# per CDNA5 ISA:
//   group0: [1:0]=count=1 | lds_addr[63:32] | global_addr[120:64] | type=2
//   group1: data_size=2B, tensor_dim0=16384, tensor_dim1=1, tile_dim0=16384,
//           tile_dim1=1, tensor_dim0_stride=16384  (1-D copy as 2D tile)
//   groups 2/3: zero (<=2D tensor)
#if USE_TDM
__device__ __forceinline__ void tdm_load_32k(unsigned int lds_off, const void* gptr) {
  const unsigned long long ga = (unsigned long long)(uintptr_t)gptr;
  u32x4 g0;
  g0[0] = 1u;                                                  // count=1 (user)
  g0[1] = lds_off;                                             // lds_addr
  g0[2] = (unsigned int)ga;                                    // global_addr lo32
  g0[3] = (unsigned int)((ga >> 32) & 0x1FFFFFFu) | (2u << 30);// ga[56:32] | type=2
  i32x8 g1;
  g1[0] = (int)(1u << 16);            // workgroup_mask=0, data_size=1 (2 bytes)
  g1[1] = (int)((16384u & 0xFFFFu) << 16);       // tensor_dim0[15:0] in [31:16]
  g1[2] = (int)((16384u >> 16) | (1u << 16));    // tensor_dim0[31:16] | tensor_dim1[15:0]=1
  g1[3] = (int)(16384u << 16);                   // tensor_dim1[31:16]=0 | tile_dim0=16384
  g1[4] = 1;                                     // tile_dim1=1, tile_dim2=0
  g1[5] = 16384;                                 // tensor_dim0_stride lo32
  g1[6] = 0;                                     // stride0 hi16 | stride1 lo16
  g1[7] = 0;                                     // stride1 hi32
  i32x4 gz = {0, 0, 0, 0};
#if __clang_major__ >= 23
  i32x8 gz8 = {0, 0, 0, 0, 0, 0, 0, 0};
  __builtin_amdgcn_tensor_load_to_lds(g0, g1, gz, gz, gz8, 0);
#else
  __builtin_amdgcn_tensor_load_to_lds(g0, g1, gz, gz, 0);
#endif
}
#endif

// ---------------------------------------------------------------------------
// WMMA fragment loaders (wave32, V_WMMA_F32_16X16X32_BF16 layouts per ISA 7.12.2)
//
// A (16x32, MxK), row-major source [rows, ld]:
//   lanes 0-15 : M=lane,    v0..v3 = K 0..7,   v4..v7 = K 16..23
//   lanes 16-31: M=lane-16, v0..v3 = K 8..15,  v4..v7 = K 24..31
__device__ __forceinline__ bf16x16 load_a_frag(const bf16_t* base, int ld,
                                               int m0, int k0, int lane) {
  const int row  = m0 + (lane & 15);
  const int koff = (lane < 16) ? 0 : 8;
  const bf16_t* p = base + (size_t)row * ld + (k0 + koff);
  FragU f;
  f.h[0] = *(const bf16x8*)(p);
  f.h[1] = *(const bf16x8*)(p + 16);
  return f.v;
}

// B (32x16, KxN), source packed as [N][Kld] (element (k,n) at n*ldk + k):
//   lanes 0-15 : N=lane,    v0..v3 = K 0..7,   v4..v7 = K 16..23
//   lanes 16-31: N=lane-16, v0..v3 = K 8..15,  v4..v7 = K 24..31
__device__ __forceinline__ bf16x16 load_b_frag(const bf16_t* base, int ldk,
                                               int k0, int n0, int lane) {
  const int col  = n0 + (lane & 15);
  const int koff = (lane < 16) ? 0 : 8;
  const bf16_t* p = base + (size_t)col * ldk + (k0 + koff);
  FragU f;
  f.h[0] = *(const bf16x8*)(p);
  f.h[1] = *(const bf16x8*)(p + 16);
  return f.v;
}

// C/D f32 16x16: VGPR j -> M = j + (lane<16?0:8), N = lane&15.

// ---------------------------------------------------------------------------
// Pack weight W[K,N] (row-major f32) -> bf16 [N][K] (B-fragment friendly)
__global__ void pack_weight_kernel(const float* __restrict__ W,
                                   bf16_t* __restrict__ out, int K, int N) {
  int idx = blockIdx.x * blockDim.x + threadIdx.x;
  if (idx >= K * N) return;
  int n = idx / K, k = idx - n * K;
  out[idx] = (bf16_t)W[(size_t)k * N + n];
}

__global__ void convert_f32_to_bf16_kernel(const float* __restrict__ x,
                                           bf16_t* __restrict__ out, int n) {
  int i = blockIdx.x * blockDim.x + threadIdx.x;
  if (i < n) out[i] = (bf16_t)x[i];
}

// ---------------------------------------------------------------------------
// xw[t,b,:] = inputs[b,t,:] @ W_x + b_x.  M=B*T=8192, K=128, N=1024.
// One 16x16 tile per wave; grid 4096 x 256 threads = 32768 waves = 512*64 tiles.
__global__ void xw_gemm_kernel(const bf16_t* __restrict__ Abf,  // [B*T, C] bf16
                               const bf16_t* __restrict__ Bpk,  // [U][C] packed
                               const float* __restrict__ b_x,
                               float* __restrict__ xw) {        // [T, B, U]
  const int warp = blockIdx.x * 8 + (threadIdx.x >> 5);
  const int lane = threadIdx.x & 31;
  const int nt = warp & 63;       // 64 N tiles
  const int mt = warp >> 6;       // 512 M tiles
  const int m0 = mt * 16, n0 = nt * 16;
  floatx8 acc = {};
#pragma unroll
  for (int k0 = 0; k0 < C_; k0 += 32) {
    bf16x16 a = load_a_frag(Abf, C_, m0, k0, lane);
    bf16x16 b = load_b_frag(Bpk, C_, k0, n0, lane);
    acc = __builtin_amdgcn_wmma_f32_16x16x32_bf16(false, a, false, b,
                                                  (short)0, acc, false, false);
  }
  const int n = n0 + (lane & 15);
  const float bv = b_x[n];
  const int mbase = m0 + ((lane >> 4) << 3);
#pragma unroll
  for (int j = 0; j < 8; ++j) {
    const int m = mbase + j;           // row in [B*T], inputs are [B,T,C]
    const int bidx = m >> 8;           // m / T_ (T_=256)
    const int tt = m & (T_ - 1);
    xw[((size_t)tt * B_ + bidx) * U_ + n] = acc[j] + bv;  // time-major
  }
}

// ---------------------------------------------------------------------------
// out = xw_t + hA @ W_h (optionally relu), write f32 and/or bf16.
// M=32, N=1024 -> 128 tiles -> grid 16 x 256 (8 waves/block, one M-tile/block).
// The 16x1024 bf16 A tile (32 KB, contiguous) is shared by all 8 waves: stage
// it into LDS once per block with the Tensor Data Mover, then feed WMMA A
// fragments from LDS (ds_load_b128) while B strips stream from global/L2.
__global__ void recur_gemm_kernel(const bf16_t* __restrict__ hA,   // [B,U] bf16
                                  const bf16_t* __restrict__ Wpk,  // [U][U] packed
                                  const float* __restrict__ xw_t,  // [B,U]
                                  float* __restrict__ out_f32,     // [B,U] or null
                                  bf16_t* __restrict__ out_bf,     // [B,U] or null
                                  int do_relu) {
  __shared__ bf16_t aTile[16 * U_];          // 32 KB
  const int wid  = threadIdx.x >> 5;
  const int lane = threadIdx.x & 31;
  const int mt = blockIdx.x >> 3;            // block-uniform M tile
  const int m0 = mt * 16;
  const int nt = (blockIdx.x & 7) * 8 + wid;
  const int n0 = nt * 16;

  const bf16_t* gA = hA + (size_t)m0 * U_;   // contiguous 32 KB
#if USE_TDM
  if (wid == 0) {                            // one wave issues the DMA
    tdm_load_32k((unsigned int)(uintptr_t)aTile, gA);
    __builtin_amdgcn_s_wait_tensorcnt(0);    // TENSORcnt is per-wave
  }
  __syncthreads();
#else
  for (int i = threadIdx.x * 8; i < 16 * U_; i += 256 * 8)
    *(bf16x8*)(aTile + i) = *(const bf16x8*)(gA + i);
  __syncthreads();
#endif

  floatx8 acc = {};
  const int arow = lane & 15;
  const int akoff = (lane < 16) ? 0 : 8;
  for (int k0 = 0; k0 < U_; k0 += 32) {
    const bf16_t* ap = aTile + arow * U_ + (k0 + akoff);
    FragU fa;
    fa.h[0] = *(const bf16x8*)(ap);          // ds_load_b128
    fa.h[1] = *(const bf16x8*)(ap + 16);
    bf16x16 b = load_b_frag(Wpk, U_, k0, n0, lane);
    acc = __builtin_amdgcn_wmma_f32_16x16x32_bf16(false, fa.v, false, b,
                                                  (short)0, acc, false, false);
  }
  const int n = n0 + (lane & 15);
  const int mbase = m0 + ((lane >> 4) << 3);
#pragma unroll
  for (int j = 0; j < 8; ++j) {
    const int m = mbase + j;
    float z = acc[j] + xw_t[m * U_ + n];
    if (do_relu) z = fmaxf(z, 0.f);
    if (out_f32) out_f32[m * U_ + n] = z;
    if (out_bf)  out_bf[m * U_ + n] = (bf16_t)z;
  }
}

// ---------------------------------------------------------------------------
// dots[b,k] = h_t[b] . h_k[b]  for k=0..t (h_t is hist slot t). One wave per (b,k).
__global__ void dots_kernel(const bf16_t* __restrict__ hist,  // [T,B,U] bf16
                            float* __restrict__ dots,          // [B,T]
                            int t) {
  const int gw = blockIdx.x * 8 + (threadIdx.x >> 5);
  const int lane = threadIdx.x & 31;
  const int nk = t + 1;
  if (gw >= B_ * nk) return;
  const int b = gw / nk;
  const int k = gw - b * nk;
  const bf16_t* hq = hist + ((size_t)t * B_ + b) * U_;
  const bf16_t* hk = hist + ((size_t)k * B_ + b) * U_;
  float s = 0.f;
  for (int u = lane * 8; u < U_; u += 32 * 8) {
    bf16x8 q = *(const bf16x8*)(hq + u);
    bf16x8 h = *(const bf16x8*)(hk + u);
#pragma unroll
    for (int i = 0; i < 8; ++i) s += (float)q[i] * (float)h[i];
  }
#pragma unroll
  for (int off = 16; off; off >>= 1) s += __shfl_xor(s, off, 32);
  if (lane == 0) dots[(size_t)b * T_ + k] = s;
}

// ---------------------------------------------------------------------------
// hs = pre + 0.5 * sum_k 0.95^(t-k) dots[b,k] * h_k ; layernorm ; relu ;
// write next-step carry in bf16. One block (256 thr) per batch row.
__global__ void fwln_kernel(const bf16_t* __restrict__ hist,   // [T,B,U]
                            const float* __restrict__ dots,    // [B,T]
                            const float* __restrict__ pre,     // [B,U]
                            const float* __restrict__ gain,
                            const float* __restrict__ bias,
                            bf16_t* __restrict__ carry_bf,     // [B,U]
                            int t) {
  const int b = blockIdx.x;
  const int tid = threadIdx.x;
  __shared__ float cbuf[T_];
  __shared__ float sred[256];
  for (int k = tid; k <= t; k += 256)
    cbuf[k] = E_LR_ * __expf(LN_L_DECAY_ * (float)(t - k)) * dots[(size_t)b * T_ + k];
  __syncthreads();

  const int v0 = tid * 4;  // 4 units per thread, U=1024
  float f0 = 0.f, f1 = 0.f, f2 = 0.f, f3 = 0.f;
  const bf16_t* hb = hist + (size_t)b * U_ + v0;
  for (int k = 0; k <= t; ++k) {
    bf16x4 hv = *(const bf16x4*)(hb + (size_t)k * (B_ * U_));
    const float c = cbuf[k];
    f0 += c * (float)hv[0];
    f1 += c * (float)hv[1];
    f2 += c * (float)hv[2];
    f3 += c * (float)hv[3];
  }
  const float* pb = pre + (size_t)b * U_ + v0;
  float hs0 = pb[0] + f0, hs1 = pb[1] + f1, hs2 = pb[2] + f2, hs3 = pb[3] + f3;

  // mean
  sred[tid] = hs0 + hs1 + hs2 + hs3;
  __syncthreads();
  for (int s = 128; s > 0; s >>= 1) { if (tid < s) sred[tid] += sred[tid + s]; __syncthreads(); }
  const float mu = sred[0] * (1.0f / U_);
  __syncthreads();
  // variance
  float d0 = hs0 - mu, d1 = hs1 - mu, d2 = hs2 - mu, d3 = hs3 - mu;
  sred[tid] = d0 * d0 + d1 * d1 + d2 * d2 + d3 * d3;
  __syncthreads();
  for (int s = 128; s > 0; s >>= 1) { if (tid < s) sred[tid] += sred[tid + s]; __syncthreads(); }
  const float inv = 1.0f / sqrtf(sred[0] * (1.0f / U_));  // reference has no epsilon

  const float* gp = gain + v0;
  const float* bp = bias + v0;
  bf16_t* cp = carry_bf + (size_t)b * U_ + v0;
  cp[0] = (bf16_t)fmaxf(gp[0] * d0 * inv + bp[0], 0.f);
  cp[1] = (bf16_t)fmaxf(gp[1] * d1 * inv + bp[1], 0.f);
  cp[2] = (bf16_t)fmaxf(gp[2] * d2 * inv + bp[2], 0.f);
  cp[3] = (bf16_t)fmaxf(gp[3] * d3 * inv + bp[3], 0.f);
}

// ---------------------------------------------------------------------------
// out = carry @ W_softmax + b_softmax. M=32, N=128, K=1024 -> 16 tiles -> grid 2.
__global__ void out_gemm_kernel(const bf16_t* __restrict__ hA,   // [B,U] bf16
                                const bf16_t* __restrict__ Wpk,  // [C][U] packed
                                const float* __restrict__ b_sm,
                                float* __restrict__ out) {       // [B,C]
  const int warp = blockIdx.x * 8 + (threadIdx.x >> 5);
  const int lane = threadIdx.x & 31;
  const int nt = warp & 7;   // 8 N tiles
  const int mt = warp >> 3;  // 2 M tiles
  const int m0 = mt * 16, n0 = nt * 16;
  floatx8 acc = {};
  for (int k0 = 0; k0 < U_; k0 += 32) {
    bf16x16 a = load_a_frag(hA, U_, m0, k0, lane);
    bf16x16 b = load_b_frag(Wpk, U_, k0, n0, lane);
    acc = __builtin_amdgcn_wmma_f32_16x16x32_bf16(false, a, false, b,
                                                  (short)0, acc, false, false);
  }
  const int n = n0 + (lane & 15);
  const float bv = b_sm[n];
  const int mbase = m0 + ((lane >> 4) << 3);
#pragma unroll
  for (int j = 0; j < 8; ++j) {
    const int m = mbase + j;
    out[m * C_ + n] = acc[j] + bv;
  }
}

// ---------------------------------------------------------------------------
extern "C" void kernel_launch(void* const* d_in, const int* in_sizes, int n_in,
                              void* d_out, int out_size, void* d_ws, size_t ws_size,
                              hipStream_t stream) {
  (void)in_sizes; (void)n_in; (void)out_size; (void)ws_size;
  const float* inputs = (const float*)d_in[0];  // [B,T,C]
  const float* W_x    = (const float*)d_in[1];  // [C,U]
  const float* b_x    = (const float*)d_in[2];  // [U]
  const float* W_h    = (const float*)d_in[3];  // [U,U]
  const float* W_sm   = (const float*)d_in[4];  // [U,C]
  const float* b_sm   = (const float*)d_in[5];  // [C]
  const float* gain   = (const float*)d_in[6];  // [U]
  const float* bias   = (const float*)d_in[7];  // [U]
  float* out = (float*)d_out;                   // [B,C]

  // Workspace carve-up (~55.3 MB total)
  char* ws = (char*)d_ws;
  size_t off = 0;
  auto carve = [&](size_t bytes) -> char* {
    char* p = ws + off;
    off = (off + bytes + 255) & ~(size_t)255;
    return p;
  };
  float*  xw       = (float*)carve((size_t)T_ * B_ * U_ * 4);   // 32 MB  [T,B,U]
  bf16_t* hist     = (bf16_t*)carve((size_t)T_ * B_ * U_ * 2);  // 16 MB  [T,B,U]
  bf16_t* in_bf    = (bf16_t*)carve((size_t)B_ * T_ * C_ * 2);  // 2 MB
  bf16_t* Wx_pack  = (bf16_t*)carve((size_t)U_ * C_ * 2);       // 256 KB [U][C]
  bf16_t* Wh_pack  = (bf16_t*)carve((size_t)U_ * U_ * 2);       // 2 MB   [U][U]
  bf16_t* Wsm_pack = (bf16_t*)carve((size_t)C_ * U_ * 2);       // 256 KB [C][U]
  float*  pre      = (float*)carve((size_t)B_ * U_ * 4);        // 128 KB
  float*  dots     = (float*)carve((size_t)B_ * T_ * 4);        // 32 KB
  bf16_t* carry_bf = (bf16_t*)carve((size_t)B_ * U_ * 2);       // 64 KB

  // One-time per call: pack weights to bf16 [N][K], convert inputs, zero carry.
  pack_weight_kernel<<<(C_ * U_ + 255) / 256, 256, 0, stream>>>(W_x, Wx_pack, C_, U_);
  pack_weight_kernel<<<(U_ * U_ + 255) / 256, 256, 0, stream>>>(W_h, Wh_pack, U_, U_);
  pack_weight_kernel<<<(U_ * C_ + 255) / 256, 256, 0, stream>>>(W_sm, Wsm_pack, U_, C_);
  convert_f32_to_bf16_kernel<<<(B_ * T_ * C_ + 255) / 256, 256, 0, stream>>>(
      inputs, in_bf, B_ * T_ * C_);
  hipMemsetAsync(carry_bf, 0, (size_t)B_ * U_ * 2, stream);

  // xw = inputs @ W_x + b_x  (time-major)
  xw_gemm_kernel<<<(512 * 64) / 8, 256, 0, stream>>>(in_bf, Wx_pack, b_x, xw);

  // Sequential scan. Per step:
  //  (1) h_t = relu(xw_t + carry @ W_h)       -> hist slot t (bf16)
  //  (2) pre = xw_t + h_t @ W_h               -> pre (f32)
  //  (3) dots[b,k] = h_t . h_k for k<=t
  //  (4) hs = pre + 0.5*sum 0.95^(t-k) dots*h_k ; LN ; relu -> carry
  for (int t = 0; t < T_; ++t) {
    const float* xw_t = xw + (size_t)t * B_ * U_;
    bf16_t* hist_t = hist + (size_t)t * B_ * U_;
    recur_gemm_kernel<<<16, 256, 0, stream>>>(carry_bf, Wh_pack, xw_t,
                                              nullptr, hist_t, 1);
    recur_gemm_kernel<<<16, 256, 0, stream>>>(hist_t, Wh_pack, xw_t,
                                              pre, nullptr, 0);
    const int nwaves = B_ * (t + 1);
    dots_kernel<<<(nwaves + 7) / 8, 256, 0, stream>>>(hist, dots, t);
    fwln_kernel<<<B_, 256, 0, stream>>>(hist, dots, pre, gain, bias, carry_bf, t);
  }

  // out = carry @ W_softmax + b_softmax
  out_gemm_kernel<<<2, 256, 0, stream>>>(carry_bf, Wsm_pack, b_sm, out);
}
